// STGumbelSaTree_764504179046
// MI455X (gfx1250) — matrix operand order repaired and checked by reading, not compiled, not run on hardware
//
#include <hip/hip_runtime.h>
#include <hip/hip_bf16.h>
#include <math.h>

#define B_   16
#define L_   48
#define WD_  1024
#define H_   1024
#define RH_  128
#define G4_  (4 * H_)
#define G6_  (6 * H_)

typedef __attribute__((ext_vector_type(16))) __bf16 v16bf;
typedef __attribute__((ext_vector_type(8)))  float  v8f;

// ---------- helpers ----------
__device__ __forceinline__ unsigned short f2bf(float f) {
    union { float f; unsigned u; } x; x.f = f;
    unsigned r = x.u + 0x7FFFu + ((x.u >> 16) & 1u);   // round-to-nearest-even
    return (unsigned short)(r >> 16);
}

__device__ __forceinline__ float sigm(float x) { return 1.0f / (1.0f + expf(-x)); }

// A-fragment (16x32 bf16, M rows on lanes 0..15): lanes 0-15 hold K={k0..k0+7, k0+16..k0+23},
// lanes 16-31 hold K={k0+8..k0+15, k0+24..k0+31}  (per CDNA5 ISA 16-bit A layout)
__device__ __forceinline__ v16bf load_frag_a(const unsigned short* __restrict__ row,
                                             int k0, int lane) {
    int koff = ((lane >> 4) & 1) * 8;
    union { v16bf v; uint4 q[2]; } u;
    u.q[0] = *(const uint4*)(row + k0 + koff);
    u.q[1] = *(const uint4*)(row + k0 + 16 + koff);
    return u.v;
}

// B-fragment (32x16 bf16, N cols on lanes 0..15): lanes 0-15 hold K={k0..k0+7, k0+16..k0+23},
// lanes 16-31 hold K={k0+8..k0+15, k0+24..k0+31} (same K order as A)
__device__ __forceinline__ v16bf load_frag_b(const unsigned short* __restrict__ row,
                                             int k0, int lane) {
    int koff = ((lane >> 4) & 1) * 8;
    union { v16bf v; uint4 q[2]; } u;
    u.q[0] = *(const uint4*)(row + k0 + koff);
    u.q[1] = *(const uint4*)(row + k0 + 16 + koff);
    return u.v;
}

// ---------- async global->LDS staging of a 16x1024 bf16 panel (32 KB) ----------
// 256 threads x 8 iters x b128 = 32768 bytes; flat row-major copy.
// Uses gfx1250 GLOBAL_LOAD_ASYNC_TO_LDS_B128 (GVS mode: saddr base + 32-bit vgpr offset),
// tracked by ASYNCcnt; low 32 bits of a generic shared pointer = LDS byte offset.
__device__ __forceinline__ void async_stage_panel(const unsigned short* __restrict__ g,
                                                  unsigned short* s, int tid) {
    unsigned lbase = (unsigned)(unsigned long long)s;
#pragma unroll
    for (int it = 0; it < 8; ++it) {
        unsigned byteoff = (unsigned)(tid * 16 + it * 4096);
        unsigned ldsa = lbase + byteoff;
        asm volatile("global_load_async_to_lds_b128 %0, %1, %2"
                     :
                     : "v"(ldsa), "v"(byteoff), "s"(g)
                     : "memory");
    }
}

__device__ __forceinline__ void async_wait_all() {
    asm volatile("s_wait_asynccnt 0x0" ::: "memory");
}

// ---------- prep kernels ----------
__global__ void cvt_bf16_kernel(const float* __restrict__ in,
                                unsigned short* __restrict__ out, int n) {
    for (int i = blockIdx.x * blockDim.x + threadIdx.x; i < n; i += gridDim.x * blockDim.x)
        out[i] = f2bf(in[i]);
}

__global__ void zero16_kernel(unsigned short* __restrict__ p, int n) {
    for (int i = blockIdx.x * blockDim.x + threadIdx.x; i < n; i += gridDim.x * blockDim.x)
        p[i] = 0;
}

__global__ void zero32_kernel(float* __restrict__ p, int n) {
    for (int i = blockIdx.x * blockDim.x + threadIdx.x; i < n; i += gridDim.x * blockDim.x)
        p[i] = 0.0f;
}

// ---------- Gx = X @ Wih^T + (bih + bhh), M=768 N=4096 K=1024, bf16 WMMA ----------
// one wave per 16x16 output tile; all 8 waves of a block share one mt -> the A panel
// (16x1024 bf16, 32 KB) is async-staged to LDS once per block and read via ds_load.
__global__ void gemm_x_kernel(const unsigned short* __restrict__ Xbf,
                              const unsigned short* __restrict__ Wihbf,
                              const float* __restrict__ bih,
                              const float* __restrict__ bhh,
                              float* __restrict__ Gx) {
    __shared__ unsigned short s_a[16 * WD_];      // 32 KB A panel
    const int tid = threadIdx.x;
    const int lane = tid & 31;
    const int wave = tid >> 5;
    const int tile = blockIdx.x * 8 + wave;       // 0..12287
    const int mt = tile >> 8;                     // 0..47 (uniform within block)
    const int nt = tile & 255;
    const int nbase = nt * 16;
    const int n = lane & 15;
    const int hi8 = (lane >> 4) * 8;

    // kick off async A-panel copy, overlap with accumulator init
    async_stage_panel(Xbf + (size_t)mt * 16 * WD_, s_a, tid);

    float bias = bih[nbase + n] + bhh[nbase + n];
    v8f acc;
#pragma unroll
    for (int v = 0; v < 8; ++v) acc[v] = bias;

    async_wait_all();
    __syncthreads();

    const unsigned short* arow = s_a + (size_t)(lane & 15) * WD_;
    const unsigned short* brow = Wihbf + (size_t)(nbase + n) * WD_;
    for (int k0 = 0; k0 < WD_; k0 += 32) {
        v16bf a = load_frag_a(arow, k0, lane);    // ds_load_b128 x2
        v16bf b = load_frag_b(brow, k0, lane);    // global b128 x2 (L2 resident)
        acc = __builtin_amdgcn_wmma_f32_16x16x32_bf16(
            false, a, false, b, (short)0, acc, false, false);
    }
#pragma unroll
    for (int v = 0; v < 8; ++v) {
        int m = mt * 16 + v + hi8;
        Gx[(size_t)m * G4_ + nbase + n] = acc[v];
    }
}

// ---------- recurrent step: G = h@Whh^T + Gx[:,t,:]; gates fused ----------
// M=16 (=batch) is one WMMA tile row. 64 waves, one per 16-wide hidden tile;
// h_{t-1} (16x1024 bf16, 32 KB) is async-staged to LDS per block (critical path).
// Each wave computes all 4 gate columns (shared A fragment from LDS), then the
// sigmoid/tanh cell update runs in-register. grid 8 x 256.
__global__ void lstm_step_kernel(const unsigned short* __restrict__ Whhbf,
                                 const float* __restrict__ Gx,
                                 const unsigned short* __restrict__ hbf_in,
                                 unsigned short* __restrict__ hbf_out,
                                 float* __restrict__ cbuf,
                                 float* __restrict__ hs,
                                 float* __restrict__ cs,
                                 int t) {
    __shared__ unsigned short s_h[16 * H_];       // 32 KB h_{t-1} panel
    const int tid = threadIdx.x;
    const int lane = tid & 31;
    const int wave = tid >> 5;
    const int ntile = blockIdx.x * 8 + wave;      // 0..63
    const int nbase = ntile * 16;
    const int n = lane & 15;
    const int hi8 = (lane >> 4) * 8;

    // start async h-panel copy; overlap with the 32 Gx accumulator loads
    async_stage_panel(hbf_in, s_h, tid);

    v8f acc[4];
#pragma unroll
    for (int g = 0; g < 4; ++g) {
        const int cg = g * H_ + nbase;
#pragma unroll
        for (int v = 0; v < 8; ++v) {
            int m = v + hi8;                      // batch row
            acc[g][v] = Gx[((size_t)m * L_ + t) * G4_ + cg + n];
        }
    }

    async_wait_all();
    __syncthreads();

    const unsigned short* arow = s_h + (size_t)(lane & 15) * H_;
    for (int k0 = 0; k0 < H_; k0 += 32) {
        v16bf a = load_frag_a(arow, k0, lane);    // ds_load_b128 x2, shared by 4 WMMAs
#pragma unroll
        for (int g = 0; g < 4; ++g) {
            const unsigned short* brow = Whhbf + (size_t)(g * H_ + nbase + n) * H_;
            __builtin_prefetch(brow + k0 + 128, 0, 1);
            v16bf b = load_frag_b(brow, k0, lane);
            acc[g] = __builtin_amdgcn_wmma_f32_16x16x32_bf16(
                false, a, false, b, (short)0, acc[g], false, false);
        }
    }

    // fused gate epilogue (each (m, j) owned by exactly one lane/element)
#pragma unroll
    for (int v = 0; v < 8; ++v) {
        int m = v + hi8;                          // batch
        int j = nbase + n;                        // hidden index
        float iv = acc[0][v], fv = acc[1][v], uv = acc[2][v], ov = acc[3][v];
        float cp = cbuf[m * H_ + j];
        float cn = sigm(fv) * cp + sigm(iv) * tanhf(uv);
        float hn = sigm(ov) * tanhf(cn);
        cbuf[m * H_ + j] = cn;
        cs[((size_t)m * L_ + t) * H_ + j] = cn;
        hs[((size_t)m * L_ + t) * H_ + j] = hn;
        hbf_out[m * H_ + j] = f2bf(hn);
    }
}

// ---------- tree phase: one block per batch, explicit post-order stack ----------
// forward-value identity: gate == one_hot(argmax(s + gumbel_noise)), so gating
// is a row selection; split pos = argmax(s). g vector staged in LDS.
__global__ void tree_kernel(const float* __restrict__ hs, const float* __restrict__ cs,
                            const float* __restrict__ gu,
                            const float* __restrict__ R1, const float* __restrict__ R2,
                            const float* __restrict__ Wc, const float* __restrict__ bc,
                            const int* __restrict__ length,
                            float* __restrict__ frames, float* __restrict__ out) {
    const int b = blockIdx.x;
    const int tid = threadIdx.x;

    __shared__ float s_g[G6_];            // 24 KB composed-gate vector
    __shared__ float s_part[256];
    __shared__ float s_scores[L_];
    __shared__ int st_start[64], st_end[64], st_phase[64], st_pos[64], st_sel[64],
                   st_parent[64], st_which[64];
    __shared__ int s_sp;

    const float* hsb = hs + (size_t)b * L_ * H_;
    const float* csb = cs + (size_t)b * L_ * H_;
    float* fbase = frames + (size_t)b * 64 * 4 * H_;  // [slot][side*2 + (h|c)][H]

    if (tid == 0) {
        int len = length[b]; if (len < 2) len = 2; if (len > L_) len = L_;
        st_start[0] = 0; st_end[0] = len; st_phase[0] = 0;
        st_parent[0] = -1; st_which[0] = 0;
        s_sp = 1;
    }
    __syncthreads();

    while (true) {
        int sp = s_sp;
        if (sp == 0) break;
        int f = sp - 1;
        int start = st_start[f], end = st_end[f], phase = st_phase[f];
        int nseg = end - start;

        if (phase == 0) {
            if (nseg <= 1) {
                int parent = st_parent[f], which = st_which[f];
                float* dh; float* dc;
                if (parent < 0) { dh = out + (size_t)b * H_; dc = out + (size_t)(B_ + b) * H_; }
                else {
                    dh = fbase + ((size_t)parent * 4 + which * 2 + 0) * H_;
                    dc = fbase + ((size_t)parent * 4 + which * 2 + 1) * H_;
                }
                if (nseg <= 0) {
                    for (int j = tid; j < H_; j += 256) { dh[j] = 0.0f; dc[j] = 0.0f; }
                } else {
                    const float* sh = hsb + (size_t)start * H_;
                    const float* sc = csb + (size_t)start * H_;
                    for (int j = tid; j < H_; j += 256) { dh[j] = sh[j]; dc[j] = sc[j]; }
                }
                __syncthreads();
                if (tid == 0) s_sp = sp - 1;
                __syncthreads();
            } else {
                // rank-MLP scores over [start,end): s = relu(hs @ R1^T) @ R2^T
                for (int p = start; p < end; ++p) {
                    const float* hrow = hsb + (size_t)p * H_;
                    int r = tid & 127, half = tid >> 7;
                    const float* w = R1 + (size_t)r * H_ + half * 512;
                    const float* x = hrow + half * 512;
                    float acc = 0.0f;
                    for (int k = 0; k < 512; ++k) acc = fmaf(w[k], x[k], acc);
                    s_part[tid] = acc;
                    __syncthreads();
                    if (tid < 128) {
                        float v = s_part[tid] + s_part[tid + 128];
                        s_part[tid] = fmaxf(v, 0.0f) * R2[tid];
                    }
                    __syncthreads();
                    for (int off = 64; off > 0; off >>= 1) {
                        if (tid < off) s_part[tid] += s_part[tid + off];
                        __syncthreads();
                    }
                    if (tid == 0) s_scores[p] = s_part[0];
                    __syncthreads();
                }
                if (tid == 0) {
                    int bp = start; float bv = -INFINITY;
                    int bs = start; float bnv = -INFINITY;
                    for (int p = start; p < end; ++p) {
                        float sv = s_scores[p];
                        if (sv > bv) { bv = sv; bp = p; }          // split = argmax(s)
                        float u = gu[(size_t)b * L_ + p];
                        float nz = -logf(-logf(u));                // Gumbel(0,1)
                        float svn = sv + nz;                       // TEMP == 1
                        if (svn > bnv) { bnv = svn; bs = p; }      // ST-gumbel hard pick
                    }
                    st_pos[f] = bp; st_sel[f] = bs; st_phase[f] = 1;
                    int ns = sp;                                   // push left child
                    st_start[ns] = start; st_end[ns] = bp; st_phase[ns] = 0;
                    st_parent[ns] = f; st_which[ns] = 0;
                    s_sp = sp + 1;
                }
                __syncthreads();
            }
        } else if (phase == 1) {
            if (tid == 0) {
                int pos = st_pos[f];
                st_phase[f] = 2;
                int ns = sp;                                       // push right child
                st_start[ns] = pos + 1; st_end[ns] = end; st_phase[ns] = 0;
                st_parent[ns] = f; st_which[ns] = 1;
                s_sp = sp + 1;
            }
            __syncthreads();
        } else {
            // phase 2: nary compose  g = [lh,rh,xh] @ Wc^T + bc
            int sel = st_sel[f];
            const float* lh = fbase + ((size_t)f * 4 + 0) * H_;
            const float* lc = fbase + ((size_t)f * 4 + 1) * H_;
            const float* rh = fbase + ((size_t)f * 4 + 2) * H_;
            const float* rc = fbase + ((size_t)f * 4 + 3) * H_;
            const float* xh = hsb + (size_t)sel * H_;
            const float* xc = csb + (size_t)sel * H_;
            for (int o = tid; o < G6_; o += 256) {
                const float* w = Wc + (size_t)o * (3 * H_);
                float acc = bc[o];
                for (int k = 0; k < H_; ++k) acc = fmaf(w[k],          lh[k], acc);
                for (int k = 0; k < H_; ++k) acc = fmaf(w[H_ + k],     rh[k], acc);
                for (int k = 0; k < H_; ++k) acc = fmaf(w[2 * H_ + k], xh[k], acc);
                s_g[o] = acc;
            }
            __syncthreads();
            int parent = st_parent[f], which = st_which[f];
            float* dh; float* dc;
            if (parent < 0) { dh = out + (size_t)b * H_; dc = out + (size_t)(B_ + b) * H_; }
            else {
                dh = fbase + ((size_t)parent * 4 + which * 2 + 0) * H_;
                dc = fbase + ((size_t)parent * 4 + which * 2 + 1) * H_;
            }
            for (int j = tid; j < H_; j += 256) {
                float gi  = s_g[j],          gfl = s_g[H_ + j],     gfr = s_g[2 * H_ + j];
                float gfx = s_g[3 * H_ + j], guv = s_g[4 * H_ + j], go  = s_g[5 * H_ + j];
                float cv = sigm(gi) * tanhf(guv) + sigm(gfl) * lc[j]
                         + sigm(gfr) * rc[j] + sigm(gfx) * xc[j];
                float hv = sigm(go) * tanhf(cv);
                dh[j] = hv; dc[j] = cv;
            }
            __syncthreads();
            if (tid == 0) s_sp = sp - 1;
            __syncthreads();
        }
    }
}

// ---------- host side ----------
extern "C" void kernel_launch(void* const* d_in, const int* in_sizes, int n_in,
                              void* d_out, int out_size, void* d_ws, size_t ws_size,
                              hipStream_t stream) {
    (void)in_sizes; (void)n_in; (void)out_size; (void)ws_size;
    const float* se  = (const float*)d_in[0];   // [B,L,WD]
    const float* gu  = (const float*)d_in[1];   // [B,L]
    const float* Wih = (const float*)d_in[2];   // [4H,WD]
    const float* Whh = (const float*)d_in[3];   // [4H,H]
    const float* bih = (const float*)d_in[4];   // [4H]
    const float* bhh = (const float*)d_in[5];   // [4H]
    const float* R1  = (const float*)d_in[6];   // [RH,H]
    const float* R2  = (const float*)d_in[7];   // [1,RH]
    const float* Wc  = (const float*)d_in[8];   // [6H,3H]
    const float* bc  = (const float*)d_in[9];   // [6H]
    const int*   len = (const int*)d_in[10];    // [B]
    float* out = (float*)d_out;                 // [2,B,H]

    char* ws = (char*)d_ws;
    size_t off = 0;
    auto alloc = [&](size_t bytes) -> void* {
        void* p = ws + off;
        off += (bytes + 255) & ~(size_t)255;
        return p;
    };
    unsigned short* Xbf   = (unsigned short*)alloc((size_t)B_ * L_ * WD_ * 2);
    unsigned short* Wihbf = (unsigned short*)alloc((size_t)G4_ * WD_ * 2);
    unsigned short* Whhbf = (unsigned short*)alloc((size_t)G4_ * H_ * 2);
    float* Gx     = (float*)alloc((size_t)B_ * L_ * G4_ * 4);
    float* hs     = (float*)alloc((size_t)B_ * L_ * H_ * 4);
    float* cs     = (float*)alloc((size_t)B_ * L_ * H_ * 4);
    unsigned short* hbf0 = (unsigned short*)alloc((size_t)B_ * H_ * 2);
    unsigned short* hbf1 = (unsigned short*)alloc((size_t)B_ * H_ * 2);
    float* cbuf   = (float*)alloc((size_t)B_ * H_ * 4);
    float* frames = (float*)alloc((size_t)B_ * 64 * 4 * H_ * 4);

    int nse = B_ * L_ * WD_;
    cvt_bf16_kernel<<<(nse + 255) / 256, 256, 0, stream>>>(se, Xbf, nse);
    int nw = G4_ * WD_;
    cvt_bf16_kernel<<<(nw + 255) / 256, 256, 0, stream>>>(Wih, Wihbf, nw);
    cvt_bf16_kernel<<<(nw + 255) / 256, 256, 0, stream>>>(Whh, Whhbf, nw);
    zero16_kernel<<<64, 256, 0, stream>>>(hbf0, B_ * H_);
    zero16_kernel<<<64, 256, 0, stream>>>(hbf1, B_ * H_);
    zero32_kernel<<<64, 256, 0, stream>>>(cbuf, B_ * H_);

    gemm_x_kernel<<<1536, 256, 0, stream>>>(Xbf, Wihbf, bih, bhh, Gx);

    for (int t = 0; t < L_; ++t) {
        const unsigned short* hin = (t & 1) ? hbf1 : hbf0;
        unsigned short* hout      = (t & 1) ? hbf0 : hbf1;
        lstm_step_kernel<<<8, 256, 0, stream>>>(Whhbf, Gx, hin, hout, cbuf, hs, cs, t);
    }

    tree_kernel<<<16, 256, 0, stream>>>(hs, cs, gu, R1, R2, Wc, bc, len, frames, out);
}
